// LorentzSegmentEncoder_87668872446025
// MI455X (gfx1250) — matrix operand order repaired and checked by reading, not compiled, not run on hardware
//
#include <hip/hip_runtime.h>
#include <hip/hip_bf16.h>

// ---------------- problem constants ----------------
#define DIN   768          // SEG_LEN*C = 24*32
#define HID   128
#define NB    512
#define NS    30           // 720/24
#define NE    4
#define NM    (NB*NS)      // 15360 rows per expert
#define EPSF  1e-6f
#define MAXNF 10.0f
#define KARCHER 10

typedef _Float16 v16h __attribute__((ext_vector_type(16)));
typedef _Float16 v8h  __attribute__((ext_vector_type(8)));
typedef _Float16 v4h  __attribute__((ext_vector_type(4)));
typedef float    v8f  __attribute__((ext_vector_type(8)));

// ---------------- wave32 reduction helpers ----------------
__device__ __forceinline__ float wredsum(float v) {
#pragma unroll
  for (int m = 16; m > 0; m >>= 1) v += __shfl_xor(v, m, 32);
  return v;
}

// =====================================================================
// Weight prep: W [E][K][H] f32 row-major  ->  Wt [E][H][K] f16 (col-major)
// =====================================================================
__global__ __launch_bounds__(256) void transpose_cast_kernel(
    const float* __restrict__ W, _Float16* __restrict__ Wt,
    int K, int H, int total)
{
  int idx = blockIdx.x * 256 + threadIdx.x;
  if (idx >= total) return;
  int e   = idx / (K * H);
  int rem = idx - e * K * H;
  int k   = rem / H;
  int h   = rem - k * H;
  Wt[(size_t)e * H * K + (size_t)h * K + k] = (_Float16)W[idx];
}

// =====================================================================
// Stage 1: per-segment normalize, tanh-scale, F = logmap0∘projx∘expmap0
// One wave per 768-wide row; 24 f32 per lane. Output f16 tangents.
// =====================================================================
__global__ __launch_bounds__(256) void stage1_kernel(
    const float* __restrict__ in0, const float* __restrict__ in1,
    const float* __restrict__ in2, const float* __restrict__ in3,
    const float* __restrict__ esc, _Float16* __restrict__ T1)
{
  const int wave = threadIdx.x >> 5;
  const int lane = threadIdx.x & 31;
  const int row  = blockIdx.x * 8 + wave;      // 0 .. NE*NM-1
  const int e    = row / NM;
  const int r    = row - e * NM;
  const float* src = (e == 0) ? in0 : (e == 1) ? in1 : (e == 2) ? in2 : in3;
  src += (size_t)r * DIN;

  float v[24];
  float s0 = 0.f, s1 = 0.f;
#pragma unroll
  for (int j = 0; j < 24; ++j) {
    v[j] = src[j * 32 + lane];
    s0 += v[j]; s1 += v[j] * v[j];
  }
  s0 = wredsum(s0); s1 = wredsum(s1);
  const float mean = s0 * (1.0f / DIN);
  const float var  = fmaxf((s1 - (float)DIN * mean * mean) * (1.0f / (DIN - 1)), 0.f);
  const float inv  = 1.0f / (sqrtf(var) + 1e-6f);
  const float scale = tanhf(esc[0]);
#pragma unroll
  for (int j = 0; j < 24; ++j) v[j] = (v[j] - mean) * inv * scale;

  // F(u): spatial = all comps except global comp 0 (= lane 0, j 0)
  float ss = 0.f;
#pragma unroll
  for (int j = 0; j < 24; ++j) ss += v[j] * v[j];
  if (lane == 0) ss -= v[0] * v[0];
  ss = wredsum(ss);
  const float n    = sqrtf(ss);
  const float nc   = fminf(fmaxf(n, EPSF), MAXNF);
  const float fac  = sinhf(nc) / fmaxf(n, EPSF);
  const float r2   = fac * fac * ss;                 // ||xs||^2
  const float x0   = sqrtf(1.0f + r2);               // projx
  const float dist = acoshf(fmaxf(x0, 1.0f + EPSF)); // logmap0
  const float g    = dist / fmaxf(sqrtf(r2), EPSF) * fac;

  _Float16* dst = T1 + (size_t)row * DIN;
#pragma unroll
  for (int j = 0; j < 24; ++j) {
    float o = g * v[j];
    if (lane == 0 && j == 0) o = 0.f;                // tangent time comp
    dst[j * 32 + lane] = (_Float16)o;
  }
}

// =====================================================================
// WMMA GEMM: Y[e] = A[e] (MxK f16, row-major) @ Wt[e]^T (Wt is HxK f16)
//            + bias[e],  Y f32 [E][M][128]
// One wave: 16 rows x 128 cols, v_wmma_f32_16x16x32_f16 over K/32 steps.
// =====================================================================
__global__ __launch_bounds__(256) void gemm_wmma_kernel(
    const _Float16* __restrict__ A,    // [E][M][K]
    const _Float16* __restrict__ Bt,   // [E][128][K]
    const float*    __restrict__ bias, // [E][128]
    float* __restrict__ Y,             // [E][M][128]
    int M, int K)
{
  const int e        = blockIdx.y;
  const int wave     = threadIdx.x >> 5;
  const int lane     = threadIdx.x & 31;
  const int row_base = blockIdx.x * 128 + wave * 16;
  if (row_base >= M) return;

  const _Float16* Ae = A  + (size_t)e * M * K;
  const _Float16* Be = Bt + (size_t)e * HID * K;

  const int arow = row_base + (lane & 15);
  const int kA   = (lane >> 4) << 3;    // 0 or 8  (A frag K phase)
  const int kB   = (lane >> 4) << 3;    // 0 or 8  (B frag K phase)

  const v8f vzero = {0.f,0.f,0.f,0.f,0.f,0.f,0.f,0.f};
  v8f acc[8];
#pragma unroll
  for (int t = 0; t < 8; ++t) acc[t] = vzero;

  for (int k0 = 0; k0 < K; k0 += 32) {
    // A fragment: 16-bit A 16x32 layout — lane holds K {kA..kA+7, kA+16..kA+23}
    const v8h* pa = (const v8h*)(Ae + (size_t)arow * K + k0 + kA);
    const v8h alo = pa[0];
    const v8h ahi = pa[2];               // +16 halves
    v16h afrag;
#pragma unroll
    for (int i = 0; i < 8; ++i) { afrag[i] = alo[i]; afrag[8 + i] = ahi[i]; }

#pragma unroll
    for (int t = 0; t < 8; ++t) {
      // B fragment: lane holds K {kB..kB+7, kB+16..kB+23} of column N
      const int col = t * 16 + (lane & 15);
      const v8h* pb = (const v8h*)(Be + (size_t)col * K + k0 + kB);
      const v8h blo = pb[0];
      const v8h bhi = pb[2];
      v16h bfrag;
#pragma unroll
      for (int i = 0; i < 8; ++i) { bfrag[i] = blo[i]; bfrag[8 + i] = bhi[i]; }
      acc[t] = __builtin_amdgcn_wmma_f32_16x16x32_f16(
          false, afrag, false, bfrag, (short)0, acc[t], false, false);
    }
  }

  // Epilogue: C layout — lane L: col = L%16, VGPR r: row = r + 8*(L/16)
  const int mh = (lane >> 4) * 8;
#pragma unroll
  for (int t = 0; t < 8; ++t) {
    const int col = t * 16 + (lane & 15);
    const float bv = bias[e * HID + col];
    float* yp = Y + ((size_t)e * M + row_base) * HID + col;
#pragma unroll
    for (int r = 0; r < 8; ++r)
      yp[(size_t)(mh + r) * HID] = acc[t][r] + bv;
  }
}

// =====================================================================
// F on a 128-dim row held as 4 floats/lane across one wave
// =====================================================================
__device__ __forceinline__ void lorentz_F4(float v[4], int lane) {
  float ss = 0.f;
#pragma unroll
  for (int i = 0; i < 4; ++i) ss += v[i] * v[i];
  if (lane == 0) ss -= v[0] * v[0];
  ss = wredsum(ss);
  const float n    = sqrtf(ss);
  const float nc   = fminf(fmaxf(n, EPSF), MAXNF);
  const float fac  = sinhf(nc) / fmaxf(n, EPSF);
  const float r2   = fac * fac * ss;
  const float x0   = sqrtf(1.0f + r2);
  const float dist = acoshf(fmaxf(x0, 1.0f + EPSF));
  const float g    = dist / fmaxf(sqrtf(r2), EPSF) * fac;
#pragma unroll
  for (int i = 0; i < 4; ++i) v[i] *= g;
  if (lane == 0) v[0] = 0.f;
}

// Between layers: t = F(F(y)), cast to f16
__global__ __launch_bounds__(256) void act_kernel(
    const float* __restrict__ Y, _Float16* __restrict__ T)
{
  const int wave = threadIdx.x >> 5, lane = threadIdx.x & 31;
  const int row  = blockIdx.x * 8 + wave;
  const float4 f = ((const float4*)(Y + (size_t)row * HID))[lane];
  float v[4] = {f.x, f.y, f.z, f.w};
  lorentz_F4(v, lane);
  lorentz_F4(v, lane);
  v4h h;
#pragma unroll
  for (int i = 0; i < 4; ++i) h[i] = (_Float16)v[i];
  *(v4h*)(T + (size_t)row * HID + lane * 4) = h;
}

// Final: z = projx(expmap0(y)) -> d_out sections [e][b][s][128]
__global__ __launch_bounds__(256) void zfinal_kernel(
    const float* __restrict__ Y, float* __restrict__ Z)
{
  const int wave = threadIdx.x >> 5, lane = threadIdx.x & 31;
  const int row  = blockIdx.x * 8 + wave;
  const float4 f = ((const float4*)(Y + (size_t)row * HID))[lane];
  float v[4] = {f.x, f.y, f.z, f.w};
  float ss = 0.f;
#pragma unroll
  for (int i = 0; i < 4; ++i) ss += v[i] * v[i];
  if (lane == 0) ss -= v[0] * v[0];
  ss = wredsum(ss);
  const float n   = sqrtf(ss);
  const float nc  = fminf(fmaxf(n, EPSF), MAXNF);
  const float fac = sinhf(nc) / fmaxf(n, EPSF);
  const float x0  = sqrtf(1.0f + fac * fac * ss);    // projx
#pragma unroll
  for (int i = 0; i < 4; ++i) v[i] *= fac;
  if (lane == 0) v[0] = x0;
  float4 o = {v[0], v[1], v[2], v[3]};
  ((float4*)(Z + (size_t)row * HID))[lane] = o;
}

// =====================================================================
// Karcher / Fréchet mean of npts 128-dim hyperboloid points, one wave.
// Points at base + s*stride (floats), lane holds comps [4*lane..4*lane+3].
// =====================================================================
__device__ void frechet_mean_wave(const float* __restrict__ base, int npts,
                                  size_t stride, int lane, float cur[4])
{
  // init: projx(expmap0(mean_s logmap0(p_s)))
  float tm[4] = {0.f, 0.f, 0.f, 0.f};
  for (int s = 0; s < npts; ++s) {
    const float4 f = ((const float4*)(base + (size_t)s * stride))[lane];
    float p[4] = {f.x, f.y, f.z, f.w};
    float ssl = 0.f;
#pragma unroll
    for (int i = 0; i < 4; ++i) ssl += p[i] * p[i];
    if (lane == 0) ssl -= p[0] * p[0];
    const float ssp  = wredsum(ssl);
    const float p0   = __shfl(p[0], 0, 32);
    const float dist = acoshf(fmaxf(p0, 1.0f + EPSF));
    const float g    = dist / fmaxf(sqrtf(ssp), EPSF);
#pragma unroll
    for (int i = 0; i < 4; ++i) tm[i] += g * p[i];
  }
  const float invN = 1.0f / (float)npts;
#pragma unroll
  for (int i = 0; i < 4; ++i) tm[i] *= invN;
  if (lane == 0) tm[0] = 0.f;
  {
    float ssl = 0.f;
#pragma unroll
    for (int i = 0; i < 4; ++i) ssl += tm[i] * tm[i];
    if (lane == 0) ssl -= tm[0] * tm[0];
    const float ss  = wredsum(ssl);
    const float n   = sqrtf(ss);
    const float nc  = fminf(fmaxf(n, EPSF), MAXNF);
    const float fac = sinhf(nc) / fmaxf(n, EPSF);
#pragma unroll
    for (int i = 0; i < 4; ++i) cur[i] = fac * tm[i];
    if (lane == 0) cur[0] = sqrtf(1.0f + fac * fac * ss);
  }
  // Karcher iterations
  for (int it = 0; it < KARCHER; ++it) {
    float gr[4] = {0.f, 0.f, 0.f, 0.f};
    for (int s = 0; s < npts; ++s) {
      const float4 f = ((const float4*)(base + (size_t)s * stride))[lane];
      float p[4] = {f.x, f.y, f.z, f.w};
      float ipl = 0.f;
#pragma unroll
      for (int i = 0; i < 4; ++i) ipl += p[i] * cur[i];
      if (lane == 0) ipl -= 2.0f * p[0] * cur[0];     // Minkowski sign
      const float ip    = wredsum(ipl);
      const float alpha = fmaxf(-ip, 1.0f + EPSF);
      const float dist  = acoshf(alpha);
      const float un    = sqrtf(fmaxf(alpha * alpha - 1.0f, 0.f));
      const float w     = dist / fmaxf(un, EPSF);
#pragma unroll
      for (int i = 0; i < 4; ++i) gr[i] += w * (p[i] + ip * cur[i]);
    }
    const float half = 0.5f * invN;                   // 0.5 * mean
#pragma unroll
    for (int i = 0; i < 4; ++i) gr[i] *= half;
    // expmap(cur, gr)
    float n2l = 0.f;
#pragma unroll
    for (int i = 0; i < 4; ++i) n2l += gr[i] * gr[i];
    if (lane == 0) n2l -= 2.0f * gr[0] * gr[0];
    const float n2  = wredsum(n2l);
    const float nn  = sqrtf(fmaxf(n2, EPSF * EPSF));
    const float ncc = fminf(nn, MAXNF);
    const float ch  = coshf(ncc);
    const float sh  = sinhf(ncc) / nn;
#pragma unroll
    for (int i = 0; i < 4; ++i) cur[i] = ch * cur[i] + sh * gr[i];
    // projx
    float ssl = 0.f;
#pragma unroll
    for (int i = 0; i < 4; ++i) ssl += cur[i] * cur[i];
    if (lane == 0) ssl -= cur[0] * cur[0];
    const float sss = wredsum(ssl);
    if (lane == 0) cur[0] = sqrtf(1.0f + sss);
  }
}

// pooled[e*B+b] = frechet over S points of z[e,b]
__global__ __launch_bounds__(256) void frechet_pool_kernel(
    const float* __restrict__ Z, float* __restrict__ pooled)
{
  const int wave = threadIdx.x >> 5, lane = threadIdx.x & 31;
  const int task = blockIdx.x * 8 + wave;            // e*NB + b
  float cur[4];
  frechet_mean_wave(Z + (size_t)task * NS * HID, NS, HID, lane, cur);
  float4 o = {cur[0], cur[1], cur[2], cur[3]};
  ((float4*)(pooled + (size_t)task * HID))[lane] = o;
}

// combined[b] = frechet over E experts; broadcast across S into out section 4
__global__ __launch_bounds__(256) void frechet_combine_kernel(
    const float* __restrict__ pooled, float* __restrict__ outc)
{
  const int wave = threadIdx.x >> 5, lane = threadIdx.x & 31;
  const int b = blockIdx.x * 8 + wave;
  float cur[4];
  frechet_mean_wave(pooled + (size_t)b * HID, NE, (size_t)NB * HID, lane, cur);
  float4 o = {cur[0], cur[1], cur[2], cur[3]};
  for (int s = 0; s < NS; ++s)
    ((float4*)(outc + ((size_t)b * NS + s) * HID))[lane] = o;
}

// =====================================================================
// Host launcher
// =====================================================================
extern "C" void kernel_launch(void* const* d_in, const int* in_sizes, int n_in,
                              void* d_out, int out_size, void* d_ws, size_t ws_size,
                              hipStream_t stream) {
  (void)in_sizes; (void)n_in; (void)out_size; (void)ws_size;
  const float* in0 = (const float*)d_in[0];
  const float* in1 = (const float*)d_in[1];
  const float* in2 = (const float*)d_in[2];
  const float* in3 = (const float*)d_in[3];
  const float* W1  = (const float*)d_in[4];
  const float* b1  = (const float*)d_in[5];
  const float* W2  = (const float*)d_in[6];
  const float* b2  = (const float*)d_in[7];
  const float* W3  = (const float*)d_in[8];
  const float* b3  = (const float*)d_in[9];
  const float* esc = (const float*)d_in[10];
  float* out = (float*)d_out;

  char* ws = (char*)d_ws;
  size_t off = 0;
  auto alloc = [&](size_t bytes) -> void* {
    void* p = ws + off;
    off = (off + bytes + 255) & ~(size_t)255;
    return p;
  };
  _Float16* t1     = (_Float16*)alloc((size_t)NE * NM * DIN * 2);   // f16 tangents L1
  float*    ybuf   = (float*)   alloc((size_t)NE * NM * HID * 4);   // f32 GEMM out
  _Float16* tbuf   = (_Float16*)alloc((size_t)NE * NM * HID * 2);   // f16 tangents L2/L3
  _Float16* Wt1    = (_Float16*)alloc((size_t)NE * DIN * HID * 2);
  _Float16* Wt2    = (_Float16*)alloc((size_t)NE * HID * HID * 2);
  _Float16* Wt3    = (_Float16*)alloc((size_t)NE * HID * HID * 2);
  float*    pooled = (float*)   alloc((size_t)NE * NB * HID * 4);

  // 0) weights -> f16 column-major
  transpose_cast_kernel<<<(NE*DIN*HID + 255)/256, 256, 0, stream>>>(W1, Wt1, DIN, HID, NE*DIN*HID);
  transpose_cast_kernel<<<(NE*HID*HID + 255)/256, 256, 0, stream>>>(W2, Wt2, HID, HID, NE*HID*HID);
  transpose_cast_kernel<<<(NE*HID*HID + 255)/256, 256, 0, stream>>>(W3, Wt3, HID, HID, NE*HID*HID);

  // 1) normalize + scale + F  -> t1
  stage1_kernel<<<NE*NM/8, 256, 0, stream>>>(in0, in1, in2, in3, esc, t1);

  // 2) layer 1 GEMM (K=768), 3) F∘F, 4) layer 2 (K=128), 5) F∘F, 6) layer 3
  gemm_wmma_kernel<<<dim3(NM/128, NE), 256, 0, stream>>>(t1,   Wt1, b1, ybuf, NM, DIN);
  act_kernel      <<<NE*NM/8, 256, 0, stream>>>(ybuf, tbuf);
  gemm_wmma_kernel<<<dim3(NM/128, NE), 256, 0, stream>>>(tbuf, Wt2, b2, ybuf, NM, HID);
  act_kernel      <<<NE*NM/8, 256, 0, stream>>>(ybuf, tbuf);
  gemm_wmma_kernel<<<dim3(NM/128, NE), 256, 0, stream>>>(tbuf, Wt3, b3, ybuf, NM, HID);

  // 7) z = projx(expmap0(y)) -> d_out sections 0..3
  zfinal_kernel<<<NE*NM/8, 256, 0, stream>>>(ybuf, out);

  // 8) Fréchet pool over S, 9) Fréchet combine over E + broadcast
  frechet_pool_kernel   <<<NE*NB/8, 256, 0, stream>>>(out, pooled);
  frechet_combine_kernel<<<NB/8,    256, 0, stream>>>(pooled, out + (size_t)NE * NB * NS * HID);
}
